// Attention_52235392254111
// MI455X (gfx1250) — compile-verified, matrix operand order checked by reading, not run on hardware
//
#include <hip/hip_runtime.h>

// ---------------------------------------------------------------------------
// Fused QKV + multi-head flash attention for MI455X (gfx1250, wave32, WMMA)
// B=2, T=2048, D=1024, H=16, hd=64.  bf16 WMMA, f32 accumulation.
// Double-buffered LDS staging via gfx1250 async global->LDS b128 copies.
// ---------------------------------------------------------------------------

typedef __attribute__((ext_vector_type(16))) __bf16 v16bf;
typedef __attribute__((ext_vector_type(8)))  float  v8f;
typedef __attribute__((ext_vector_type(4)))  int    v4i;

union V8   { v8f v; float f[8]; };
union Frag { uint4 q[2]; v16bf v; };

__device__ __forceinline__ unsigned bfbits(float f) {
  union { float f; unsigned u; } x; x.f = f;
  return (x.u + 0x7fffu + ((x.u >> 16) & 1u)) >> 16;   // RNE
}
__device__ __forceinline__ __bf16 f2bf(float f) {
  unsigned short h = (unsigned short)bfbits(f);
  return __builtin_bit_cast(__bf16, h);
}
__device__ __forceinline__ unsigned pack2(float lo, float hi) {
  return bfbits(lo) | (bfbits(hi) << 16);
}

// ---- gfx1250 async global->LDS copy (ASYNCcnt) with portable fallback -----
// Builtin signature (from hipcc diagnostic): (v4i AS1*, v4i AS3*, imm, imm)
__device__ __forceinline__ void async_cp16(__bf16* lds, const __bf16* g) {
#if __has_builtin(__builtin_amdgcn_global_load_async_to_lds_b128)
  __builtin_amdgcn_global_load_async_to_lds_b128(
      (__attribute__((address_space(1))) v4i*)(g),
      (__attribute__((address_space(3))) v4i*)(lds), 0, 0);
#else
  *(uint4*)lds = *(const uint4*)g;
#endif
}
template <int N>
__device__ __forceinline__ void wait_async() {
#if __has_builtin(__builtin_amdgcn_s_wait_asynccnt)
  __builtin_amdgcn_s_wait_asynccnt(N);
#endif
}

// 16-bit A/B fragment layout (ISA 7.12.2): lane half = lane[4];
// elements 0..7  -> K = base + half*8 + e        (16B contiguous)
// elements 8..15 -> K = base + 16 + half*8 + e-8 (16B contiguous)
__device__ __forceinline__ v16bf load_frag16(const __bf16* p) {
  Frag u;
  u.q[0] = *(const uint4*)(p);
  u.q[1] = *(const uint4*)(p + 16);
  return u.v;
}

// ---------------------------------------------------------------------------
// Kernel 1a: x f32 -> bf16, vectorized (8 elements / thread / iter)
// ---------------------------------------------------------------------------
__global__ void cvt_x(const float4* __restrict__ in, uint4* __restrict__ out, int n8) {
  for (int i = blockIdx.x * blockDim.x + threadIdx.x; i < n8; i += gridDim.x * blockDim.x) {
    float4 a = in[2 * i], b = in[2 * i + 1];
    uint4 o;
    o.x = pack2(a.x, a.y); o.y = pack2(a.z, a.w);
    o.z = pack2(b.x, b.y); o.w = pack2(b.z, b.w);
    out[i] = o;
  }
}

// ---------------------------------------------------------------------------
// Kernel 1b: W f32 [K=1024, N=3072] -> bf16 W^T [N=3072, K=1024]
// (k-contiguous so GEMM B staging is a straight b128 async copy)
// ---------------------------------------------------------------------------
__global__ void cvt_wT(const float* __restrict__ W, __bf16* __restrict__ Wt) {
  for (int i = blockIdx.x * blockDim.x + threadIdx.x; i < 1024 * 3072;
       i += gridDim.x * blockDim.x) {
    const int k = i / 3072, n = i - k * 3072;
    Wt[(long)n * 1024 + k] = f2bf(W[i]);
  }
}

// ---------------------------------------------------------------------------
// Kernel 2: QKV GEMM.  M=4096 (B*T), N=3072, K=1024.
// 256 thr = 8 waves; 128x128 tile, K-step 64, double-buffered async staging.
// Q scaled by 1/8 -> [B,H,T,hd]; K -> [B,H,T,hd]; V -> [B,H,hd,T] (pre-
// transposed so flash V staging is contiguous).
// ---------------------------------------------------------------------------
__global__ __launch_bounds__(256) void qkv_gemm(
    const __bf16* __restrict__ xb, const __bf16* __restrict__ Wt,
    const float* __restrict__ bias,
    __bf16* __restrict__ Qb, __bf16* __restrict__ Kb, __bf16* __restrict__ Vb) {
  __shared__ __bf16 As[2][128 * 64];   // As[m][k], k contiguous
  __shared__ __bf16 Bs[2][128 * 64];   // Bs[n][k], k contiguous (from W^T)

  const int tid   = threadIdx.x;
  const int lane  = tid & 31;
  const int w     = tid >> 5;
  const int lanen = lane & 15;
  const int khalf = (lane >> 4) & 1;
  const int wm    = w >> 2;          // 0..1
  const int wn    = w & 3;           // 0..3
  const int Mbase = blockIdx.x * 128;
  const int Nbase = blockIdx.y * 128;

  V8 acc[4][2];
  #pragma unroll
  for (int mi = 0; mi < 4; ++mi)
    #pragma unroll
    for (int ni = 0; ni < 2; ++ni)
      #pragma unroll
      for (int r = 0; r < 8; ++r) acc[mi][ni].f[r] = 0.f;

  // stage one 128x64 A tile + 128x64 B tile: 8 async b128 per thread
  auto stage = [&](int buf, int k0) {
    #pragma unroll
    for (int j = 0; j < 4; ++j) {               // 1024 chunks of 8 bf16
      const int i = tid + j * 256;
      const int row = i >> 3, seg = (i & 7) * 8;
      async_cp16(&As[buf][row * 64 + seg], &xb[(long)(Mbase + row) * 1024 + k0 + seg]);
    }
    #pragma unroll
    for (int j = 0; j < 4; ++j) {
      const int i = tid + j * 256;
      const int row = i >> 3, seg = (i & 7) * 8;
      async_cp16(&Bs[buf][row * 64 + seg], &Wt[(long)(Nbase + row) * 1024 + k0 + seg]);
    }
  };

  stage(0, 0);
  for (int it = 0; it < 16; ++it) {
    const int cur = it & 1;
    if (it < 15) { stage(1 - cur, (it + 1) * 64); wait_async<8>(); }
    else         { wait_async<0>(); }
    __syncthreads();

    #pragma unroll
    for (int c = 0; c < 2; ++c) {               // two K=32 chunks
      v16bf bfrag[2];
      #pragma unroll
      for (int ni = 0; ni < 2; ++ni)
        bfrag[ni] = load_frag16(&Bs[cur][(wn * 32 + ni * 16 + lanen) * 64 + c * 32 + khalf * 8]);
      #pragma unroll
      for (int mi = 0; mi < 4; ++mi) {
        v16bf afrag = load_frag16(&As[cur][(wm * 64 + mi * 16 + lanen) * 64 + c * 32 + khalf * 8]);
        #pragma unroll
        for (int ni = 0; ni < 2; ++ni)
          acc[mi][ni].v = __builtin_amdgcn_wmma_f32_16x16x32_bf16(
              false, afrag, false, bfrag[ni], (short)0, acc[mi][ni].v, false, false);
      }
    }
    __syncthreads();
  }

  // epilogue: +bias, split Q/K/V, fold 1/sqrt(64) into Q, V stored [bh][hd][t]
  #pragma unroll
  for (int mi = 0; mi < 4; ++mi) {
    #pragma unroll
    for (int ni = 0; ni < 2; ++ni) {
      #pragma unroll
      for (int r = 0; r < 8; ++r) {
        const int m = Mbase + wm * 64 + mi * 16 + r + khalf * 8;   // 0..4095
        const int n = Nbase + wn * 32 + ni * 16 + lanen;           // 0..3071
        const float val = acc[mi][ni].f[r] + bias[n];
        const int sec = n >> 10, d = n & 1023;
        const int h = d >> 6, hd = d & 63;
        const int bb = m >> 11, t = m & 2047;
        const int bh = bb * 16 + h;
        if      (sec == 0) Qb[((long)bh * 2048 + t) * 64 + hd] = f2bf(val * 0.125f);
        else if (sec == 1) Kb[((long)bh * 2048 + t) * 64 + hd] = f2bf(val);
        else               Vb[((long)bh * 64 + hd) * 2048 + t] = f2bf(val);
      }
    }
  }
}

// ---------------------------------------------------------------------------
// Kernel 3: flash attention.  WG = 4 waves / 64 q-rows of one (b,h); 32 key
// blocks of 64 streamed through double-buffered LDS via async b128 copies.
// ---------------------------------------------------------------------------
__global__ __launch_bounds__(128) void flash_attn(
    const __bf16* __restrict__ Qb, const __bf16* __restrict__ Kb,
    const __bf16* __restrict__ Vb, float* __restrict__ out) {
  __shared__ __bf16 Ks[2][64 * 64];    // Ks[key][hd]  (hd contiguous)
  __shared__ __bf16 Vt[2][64 * 64];    // Vt[hd][key]  (key contiguous)
  __shared__ __bf16 Ps[4][16 * 64];    // per-wave P tile [qrow][key]

  const int tid   = threadIdx.x;
  const int lane  = tid & 31;
  const int w     = tid >> 5;
  const int lanen = lane & 15;
  const int khalf = (lane >> 4) & 1;
  const int bh    = blockIdx.y;
  const int qbase = blockIdx.x * 64;

  const __bf16* Qh = Qb + (long)bh * 2048 * 64;
  const __bf16* Kh = Kb + (long)bh * 2048 * 64;
  const __bf16* Vh = Vb + (long)bh * 64 * 2048;   // [hd][t]

  // stage K block (direct) + V block (already transposed in global layout)
  auto stage = [&](int buf, int kb) {
    const __bf16* Kblk = Kh + (long)kb * 64 * 64;
    #pragma unroll
    for (int j = 0; j < 4; ++j) {               // 512 chunks of 8 bf16
      const int i = tid + j * 128;
      async_cp16(&Ks[buf][i * 8], Kblk + i * 8);
    }
    #pragma unroll
    for (int j = 0; j < 4; ++j) {
      const int i = tid + j * 128;
      const int hd = i >> 3, seg = (i & 7) * 8;
      async_cp16(&Vt[buf][hd * 64 + seg], &Vh[(long)hd * 2048 + kb * 64 + seg]);
    }
  };

  // Q fragments for this wave's 16 rows
  v16bf aq[2];
  {
    const __bf16* qrow = Qh + (long)(qbase + w * 16 + lanen) * 64;
    #pragma unroll
    for (int c = 0; c < 2; ++c) aq[c] = load_frag16(qrow + c * 32 + khalf * 8);
  }

  V8 O[4];
  float rmax[8], rsum[8];
  #pragma unroll
  for (int g = 0; g < 4; ++g)
    #pragma unroll
    for (int r = 0; r < 8; ++r) O[g].f[r] = 0.f;
  #pragma unroll
  for (int r = 0; r < 8; ++r) { rmax[r] = -1e30f; rsum[r] = 0.f; }

  stage(0, 0);
  for (int kb = 0; kb < 32; ++kb) {
    const int cur = kb & 1;
    if (kb < 31) { stage(1 - cur, kb + 1); wait_async<8>(); }
    else         { wait_async<0>(); }
    __syncthreads();

    // S = Q K^T : 4 key-groups x 2 K-chunks
    V8 S[4];
    #pragma unroll
    for (int g = 0; g < 4; ++g) {
      #pragma unroll
      for (int r = 0; r < 8; ++r) S[g].f[r] = 0.f;
      #pragma unroll
      for (int c = 0; c < 2; ++c) {
        v16bf bk = load_frag16(&Ks[cur][(g * 16 + lanen) * 64 + c * 32 + khalf * 8]);
        S[g].v = __builtin_amdgcn_wmma_f32_16x16x32_bf16(
            false, aq[c], false, bk, (short)0, S[g].v, false, false);
      }
    }

    // online softmax (C-layout: VGPR r <-> rows r / r+8, lanes span N)
    float mnew[8], alpha[8];
    #pragma unroll
    for (int r = 0; r < 8; ++r) {
      float mb = fmaxf(fmaxf(S[0].f[r], S[1].f[r]), fmaxf(S[2].f[r], S[3].f[r]));
      mb = fmaxf(mb, __shfl_xor(mb, 8));
      mb = fmaxf(mb, __shfl_xor(mb, 4));
      mb = fmaxf(mb, __shfl_xor(mb, 2));
      mb = fmaxf(mb, __shfl_xor(mb, 1));
      mnew[r]  = fmaxf(rmax[r], mb);
      alpha[r] = __expf(rmax[r] - mnew[r]);
      rmax[r]  = mnew[r];
    }

    float lb[8];
    #pragma unroll
    for (int r = 0; r < 8; ++r) lb[r] = 0.f;
    #pragma unroll
    for (int g = 0; g < 4; ++g)
      #pragma unroll
      for (int r = 0; r < 8; ++r) {
        const float p = __expf(S[g].f[r] - mnew[r]);
        lb[r] += p;
        Ps[w][(r + khalf * 8) * 64 + g * 16 + lanen] = f2bf(p);
      }
    #pragma unroll
    for (int r = 0; r < 8; ++r) {
      lb[r] += __shfl_xor(lb[r], 8);
      lb[r] += __shfl_xor(lb[r], 4);
      lb[r] += __shfl_xor(lb[r], 2);
      lb[r] += __shfl_xor(lb[r], 1);
      rsum[r] = rsum[r] * alpha[r] + lb[r];
    }
    #pragma unroll
    for (int g = 0; g < 4; ++g)
      #pragma unroll
      for (int r = 0; r < 8; ++r) O[g].f[r] *= alpha[r];

    // O += P V (per-wave LDS region; same-wave LDS ops are in-order)
    v16bf ap[2];
    #pragma unroll
    for (int c = 0; c < 2; ++c)
      ap[c] = load_frag16(&Ps[w][lanen * 64 + c * 32 + khalf * 8]);
    #pragma unroll
    for (int g = 0; g < 4; ++g)
      #pragma unroll
      for (int c = 0; c < 2; ++c) {
        v16bf bv = load_frag16(&Vt[cur][(g * 16 + lanen) * 64 + c * 32 + khalf * 8]);
        O[g].v = __builtin_amdgcn_wmma_f32_16x16x32_bf16(
            false, ap[c], false, bv, (short)0, O[g].v, false, false);
      }
    __syncthreads();
  }

  // normalize, write y[b, t, h*64+hd] f32
  const int b = bh >> 4, h = bh & 15;
  #pragma unroll
  for (int g = 0; g < 4; ++g)
    #pragma unroll
    for (int r = 0; r < 8; ++r) {
      const int t  = qbase + w * 16 + r + khalf * 8;
      const int hd = g * 16 + lanen;
      out[((long)(b * 2048 + t)) * 1024 + h * 64 + hd] = O[g].f[r] / rsum[r];
    }
}

// ---------------------------------------------------------------------------
// Launch
// ---------------------------------------------------------------------------
extern "C" void kernel_launch(void* const* d_in, const int* in_sizes, int n_in,
                              void* d_out, int out_size, void* d_ws, size_t ws_size,
                              hipStream_t stream) {
  const float* x    = (const float*)d_in[0];   // [2,2048,1024]
  const float* W    = (const float*)d_in[1];   // [1024,3072]
  const float* bias = (const float*)d_in[2];   // [3072]
  float* out = (float*)d_out;                  // [2,2048,1024]

  // workspace (bf16): W^T | x | Q | K | V
  __bf16* Wt = (__bf16*)d_ws;
  __bf16* xb = Wt + (size_t)3072 * 1024;
  __bf16* Qb = xb + (size_t)4096 * 1024;
  __bf16* Kb = Qb + (size_t)32 * 2048 * 64;
  __bf16* Vb = Kb + (size_t)32 * 2048 * 64;

  cvt_x<<<2048, 256, 0, stream>>>((const float4*)x, (uint4*)xb, 4096 * 1024 / 8);
  cvt_wT<<<4096, 256, 0, stream>>>(W, Wt);

  qkv_gemm<<<dim3(32, 24), 256, 0, stream>>>(xb, Wt, bias, Qb, Kb, Vb);

  flash_attn<<<dim3(32, 32), 128, 0, stream>>>(Qb, Kb, Vb, out);
}
